// GIN_79328045957731
// MI455X (gfx1250) — matrix operand order repaired and checked by reading, not compiled, not run on hardware
//
#include <hip/hip_runtime.h>
#include <hip/hip_bf16.h>
#include <stdint.h>

typedef __attribute__((ext_vector_type(4)))  __bf16 v4bf;
typedef __attribute__((ext_vector_type(8)))  __bf16 v8bf;
typedef __attribute__((ext_vector_type(16))) __bf16 v16bf;
typedef __attribute__((ext_vector_type(8)))  float  v8f;

#define NNODES 50000
#define HIDDIM 256
#define NCLS   6

// ---------------- zero ----------------
__global__ void k_zero(float* __restrict__ p, long n) {
    long i = (long)blockIdx.x * blockDim.x + threadIdx.x;
    if (i < n) p[i] = 0.0f;
}

// ---------------- edge scatter-add: agg[dst] += h[src] ----------------
// thread t -> edge e = t >> lc, feature chunk c = (t & (chunks-1))*4
__global__ void k_scatter(const float* __restrict__ h, const int* __restrict__ src,
                          const int* __restrict__ dst, float* __restrict__ agg,
                          long total, int D, int lc) {
    long t = (long)blockIdx.x * blockDim.x + threadIdx.x;
    if (t >= total) return;
    long e = t >> lc;
    int  c = (int)(t & ((1 << lc) - 1)) << 2;
    int  s = src[e], d = dst[e];
    const float4 v = *(const float4*)(h + (long)s * D + c);
    float* o = agg + (long)d * D + c;
    atomicAdd(o + 0, v.x);
    atomicAdd(o + 1, v.y);
    atomicAdd(o + 2, v.z);
    atomicAdd(o + 3, v.w);
}

// ---------------- z = bf16(a + b), 4 elements per thread ----------------
__global__ void k_addcvt(const float* __restrict__ a, const float* __restrict__ b,
                         __bf16* __restrict__ z, long n4) {
    long i = (long)blockIdx.x * blockDim.x + threadIdx.x;
    if (i >= n4) return;
    long i4 = i << 2;
    float4 va = *(const float4*)(a + i4);
    float4 vb = *(const float4*)(b + i4);
    v4bf o;
    o[0] = (__bf16)(va.x + vb.x);
    o[1] = (__bf16)(va.y + vb.y);
    o[2] = (__bf16)(va.z + vb.z);
    o[3] = (__bf16)(va.w + vb.w);
    *(v4bf*)(z + i4) = o;
}

// ---------------- pack W[K][NC] f32 -> per-lane bf16 B fragments ----------------
// fragment layout (ISA 7.12.2, 16-bit B 32x16): lane l: n = l&15, khalf = l>>4,
// elems 0..7 = W[kt*32 + khalf*8 + e][n], elems 8..15 = W[kt*32 + 16 + khalf*8 + e][n]
__global__ void k_pack(const float* __restrict__ W, __bf16* __restrict__ P, int K, int NC) {
    int t = blockIdx.x * blockDim.x + threadIdx.x;
    int NT = NC >> 4;
    int total = (K >> 5) * NT * 32;
    if (t >= total) return;
    int lane = t & 31;
    int tile = t >> 5;
    int nt = tile % NT;
    int kt = tile / NT;
    int n = nt * 16 + (lane & 15);
    int kbase = kt * 32 + (lane >> 4) * 8;
    __bf16* o = P + (long)t * 16;
    #pragma unroll
    for (int e = 0; e < 8; ++e) {
        o[e]     = (__bf16)W[(long)(kbase + e) * NC + n];
        o[e + 8] = (__bf16)W[(long)(kbase + 16 + e) * NC + n];
    }
}

// build 16-bit A/B fragment from two contiguous 8-element chunks (k, k+16)
__device__ __forceinline__ v16bf load_frag(const __bf16* __restrict__ p) {
    v8bf lo = *(const v8bf*)(p);
    v8bf hi = *(const v8bf*)(p + 16);
    return __builtin_shufflevector(lo, hi, 0, 1, 2, 3, 4, 5, 6, 7,
                                   8, 9, 10, 11, 12, 13, 14, 15);
}

// ---------------- GEMM: out = relu(A[nrows x K](bf16) @ W + bias), NC = 256 ----------------
// 256 threads = 8 waves; wave -> (rowTile = 2*blk + w&1, colGroup = w>>1): 16 rows x 64 cols,
// 4 accumulator tiles/wave. K is compile-time: fully unrolled, 2-stage software pipeline so
// the next k-tile's global_load_b128 clauses overlap the current 4 WMMAs.
template <bool OUT_BF16, int K>
__global__ void __launch_bounds__(256)
k_gemm(const __bf16* __restrict__ A, const __bf16* __restrict__ Bp,
       const float* __restrict__ bias, void* __restrict__ outv, int nrows) {
    const int NC = 256;
    const int NT = NC / 16;            // 16 col tiles
    const int KT = K / 32;             // k-tiles
    int lane  = threadIdx.x & 31;
    int wave  = threadIdx.x >> 5;
    int rowBase  = (blockIdx.x * 2 + (wave & 1)) * 16;
    int colGroup = wave >> 1;          // 0..3
    int m     = lane & 15;
    int khalf = lane >> 4;

    long rowA = rowBase + m;
    if (rowA >= nrows) rowA = nrows - 1;              // clamp loads for tail tile
    const __bf16* arow = A + rowA * (long)K + khalf * 8;          // + kt*32
    const __bf16* brow = Bp + ((long)colGroup * 4 * 32 + lane) * 16; // + kt*NT*512 + t*512

    v8f acc[4] = {};

    v16bf a_cur = load_frag(arow);
    v16bf b_cur[4];
    #pragma unroll
    for (int t = 0; t < 4; ++t) b_cur[t] = *(const v16bf*)(brow + t * 512);

    #pragma unroll
    for (int kt = 0; kt < KT; ++kt) {
        v16bf a_nxt, b_nxt[4];
        if (kt + 1 < KT) {
            a_nxt = load_frag(arow + (kt + 1) * 32);
            #pragma unroll
            for (int t = 0; t < 4; ++t)
                b_nxt[t] = *(const v16bf*)(brow + (long)(kt + 1) * NT * 512 + t * 512);
        }
        #pragma unroll
        for (int t = 0; t < 4; ++t)
            acc[t] = __builtin_amdgcn_wmma_f32_16x16x32_bf16(
                false, a_cur, false, b_cur[t], (short)0, acc[t], false, false);
        if (kt + 1 < KT) {
            a_cur = a_nxt;
            #pragma unroll
            for (int t = 0; t < 4; ++t) b_cur[t] = b_nxt[t];
        }
    }

    // epilogue: C VGPR r -> row = rowBase + r + khalf*8, col = colBase + (lane&15)
    #pragma unroll
    for (int t = 0; t < 4; ++t) {
        int col = (colGroup * 4 + t) * 16 + (lane & 15);
        float bv = bias[col];
        #pragma unroll
        for (int r = 0; r < 8; ++r) {
            int row = rowBase + r + khalf * 8;
            if (row < nrows) {
                float v = acc[t][r] + bv;
                v = v > 0.0f ? v : 0.0f;
                if (OUT_BF16)
                    ((__bf16*)outv)[(long)row * NC + col] = (__bf16)v;
                else
                    ((float*)outv)[(long)row * NC + col] = v;
            }
        }
    }
}

// ---------------- column sum over nodes: s[256] += sum_r h[r][:] ----------------
__global__ void k_colsum(const float* __restrict__ h, float* __restrict__ s, int n) {
    int col = threadIdx.x;   // 256 threads
    float acc = 0.0f;
    for (int r = blockIdx.x; r < n; r += gridDim.x)
        acc += h[(long)r * HIDDIM + col];
    atomicAdd(&s[col], acc);
}

// ---------------- readout: out[c] = colsum @ lw[:,c] + N*lb[c] ----------------
__global__ void k_final(const float* __restrict__ s, const float* __restrict__ lw,
                        const float* __restrict__ lb, float* __restrict__ out) {
    int c = threadIdx.x;
    if (c < NCLS) {
        float acc = (float)NNODES * lb[c];
        for (int k = 0; k < HIDDIM; ++k)
            acc += s[k] * lw[k * NCLS + c];
        out[c] = acc;
    }
}

extern "C" void kernel_launch(void* const* d_in, const int* in_sizes, int n_in,
                              void* d_out, int out_size, void* d_ws, size_t ws_size,
                              hipStream_t stream) {
    const float* x   = (const float*)d_in[0];
    const int*   ei  = (const int*)d_in[1];
    const float* w0a = (const float*)d_in[2];
    const float* b0a = (const float*)d_in[3];
    const float* w0b = (const float*)d_in[4];
    const float* b0b = (const float*)d_in[5];
    const float* w1a = (const float*)d_in[6];
    const float* b1a = (const float*)d_in[7];
    const float* w1b = (const float*)d_in[8];
    const float* b1b = (const float*)d_in[9];
    const float* lw  = (const float*)d_in[10];
    const float* lb  = (const float*)d_in[11];

    const int nE = in_sizes[1] / 2;
    const int* src = ei;
    const int* dst = ei + nE;

    char* ws = (char*)d_ws;
    float*  agg = (float*)(ws + 0);              // 50000*256*4 = 51,200,000 (also reused for tbf)
    __bf16* tbf = (__bf16*)(ws + 0);             // bf16 intermediate (25.6MB, fits in agg space)
    __bf16* zbf = (__bf16*)(ws + 51200000);      // 25,600,000
    float*  h   = (float*)(ws + 76800000);       // 51,200,000
    __bf16* pw  = (__bf16*)(ws + 128000000);     // 131,072 (packed weights, reused per GEMM)
    float*  cs  = (float*)(ws + 128131072);      // 1,024

    const int nGemmBlk = (NNODES + 31) / 32;     // 1563

    // ---------------- layer 0 (D = 128) ----------------
    {
        long n = (long)NNODES * 128;
        k_zero<<<(unsigned)((n + 255) / 256), 256, 0, stream>>>(agg, n);
        long tot = (long)nE * 32;                                    // 128/4 chunks
        k_scatter<<<(unsigned)((tot + 255) / 256), 256, 0, stream>>>(x, src, dst, agg, tot, 128, 5);
        long n4 = n / 4;
        k_addcvt<<<(unsigned)((n4 + 255) / 256), 256, 0, stream>>>(x, agg, zbf, n4);
        k_pack<<<8, 256, 0, stream>>>(w0a, pw, 128, 256);
        k_gemm<true, 128><<<nGemmBlk, 256, 0, stream>>>(zbf, pw, b0a, (void*)tbf, NNODES);
        k_pack<<<16, 256, 0, stream>>>(w0b, pw, 256, 256);
        k_gemm<false, 256><<<nGemmBlk, 256, 0, stream>>>(tbf, pw, b0b, (void*)h, NNODES);
    }

    // ---------------- layer 1 (D = 256) ----------------
    {
        long n = (long)NNODES * 256;
        k_zero<<<(unsigned)((n + 255) / 256), 256, 0, stream>>>(agg, n);
        long tot = (long)nE * 64;                                    // 256/4 chunks
        k_scatter<<<(unsigned)((tot + 255) / 256), 256, 0, stream>>>(h, src, dst, agg, tot, 256, 6);
        long n4 = n / 4;
        k_addcvt<<<(unsigned)((n4 + 255) / 256), 256, 0, stream>>>(h, agg, zbf, n4);
        k_pack<<<16, 256, 0, stream>>>(w1a, pw, 256, 256);
        k_gemm<true, 256><<<nGemmBlk, 256, 0, stream>>>(zbf, pw, b1a, (void*)tbf, NNODES);
        k_pack<<<16, 256, 0, stream>>>(w1b, pw, 256, 256);
        k_gemm<false, 256><<<nGemmBlk, 256, 0, stream>>>(tbf, pw, b1b, (void*)h, NNODES);
    }

    // ---------------- readout ----------------
    k_zero<<<1, 256, 0, stream>>>(cs, 256);
    k_colsum<<<256, 256, 0, stream>>>(h, cs, NNODES);
    k_final<<<1, 64, 0, stream>>>(cs, lw, lb, (float*)d_out);
}
